// LadderNetwork_31516470018353
// MI455X (gfx1250) — matrix operand order repaired and checked by reading, not compiled, not run on hardware
//
#include <hip/hip_runtime.h>
#include <stdint.h>
#include <stddef.h>

// ---------------------------------------------------------------------------
// Ladder Network forward for MI455X (gfx1250, wave32, WMMA 16x16x32 f16).
// GEMM staging uses GLOBAL_LOAD_ASYNC_TO_LDS_B128 (ASYNCcnt) to avoid the
// per-chunk global_load -> s_wait_loadcnt 0 -> ds_store serialization.
// ---------------------------------------------------------------------------

typedef __attribute__((ext_vector_type(16))) _Float16 v16h;
typedef __attribute__((ext_vector_type(8)))  _Float16 v8h;
typedef __attribute__((ext_vector_type(8)))  float    v8f;
typedef int v4i_g __attribute__((vector_size(16)));   // GCC vector, matches builtin

#define MROWS 16384
#define TM 128
#define TN 64
#define TK 32

#if defined(__gfx1250__) && \
    __has_builtin(__builtin_amdgcn_global_load_async_to_lds_b128)
#define USE_ASYNC_LDS 1
#else
#define USE_ASYNC_LDS 0
#endif

#if USE_ASYNC_LDS
__device__ __forceinline__ void async_copy16(const _Float16* g, _Float16* l)
{
    // 16B global -> LDS, no VGPR round trip, tracked by ASYNCcnt.
    __builtin_amdgcn_global_load_async_to_lds_b128(
        (__attribute__((address_space(1))) v4i_g*)(void*)g,
        (__attribute__((address_space(3))) v4i_g*)(void*)l,
        /*imm offset=*/0, /*cpol=*/0);
}
__device__ __forceinline__ void wait_async0()
{
#if __has_builtin(__builtin_amdgcn_s_wait_asynccnt)
    __builtin_amdgcn_s_wait_asynccnt(0);
#else
    asm volatile("s_wait_asynccnt 0x0" ::: "memory");
#endif
}
#endif

union V16 { v16h v; v8h h[2]; };

// ------------------------------ GEMM (WMMA) --------------------------------
// C[M][Npad] = A[M][Kpad](f16) * W[Npad][Kpad]^T(f16) + bias   (f32 accum)
// 256 threads = 8 wave32 waves; wave (wm,wn) computes a 32x32 tile of the
// 128x64 block tile. Double-buffered LDS staging, K-step 32.
__global__ __launch_bounds__(256)
void gemm_wmma_f16(const _Float16* __restrict__ A,
                   const _Float16* __restrict__ W,
                   const float* __restrict__ bias, int biasN,
                   float* __restrict__ C,
                   int Kpad, int Npad)
{
    __shared__ _Float16 As[2][TM][TK];
    __shared__ _Float16 Ws[2][TN][TK];

    const int tid  = threadIdx.x;
    const int lane = tid & 31;
    const int wave = tid >> 5;
    const int wm   = wave & 3;      // M quadrant (0..3)
    const int wn   = wave >> 2;     // N half     (0..1)
    const int mblk = blockIdx.x * TM;
    const int nblk = blockIdx.y * TN;
    const int r    = lane & 15;     // row (A) / col (B) within 16
    const int kg   = lane >> 4;     // lane-half K group

    v8f acc[2][2] = {};

    const int ktiles = Kpad / TK;

    // ---- stage tile 0 ----
    {
#if USE_ASYNC_LDS
#pragma unroll
        for (int c = tid; c < (TM * TK / 8); c += 256) {
            const int row = c >> 2, col = (c & 3) * 8;
            async_copy16(&A[(size_t)(mblk + row) * Kpad + col], &As[0][row][col]);
        }
        {
            const int row = tid >> 2, col = (tid & 3) * 8;
            async_copy16(&W[(size_t)(nblk + row) * Kpad + col], &Ws[0][row][col]);
        }
#else
#pragma unroll
        for (int c = tid; c < (TM * TK / 8); c += 256) {
            const int row = c >> 2, col = (c & 3) * 8;
            *(v8h*)&As[0][row][col] =
                *(const v8h*)&A[(size_t)(mblk + row) * Kpad + col];
        }
        {
            const int row = tid >> 2, col = (tid & 3) * 8;
            *(v8h*)&Ws[0][row][col] =
                *(const v8h*)&W[(size_t)(nblk + row) * Kpad + col];
        }
#endif
    }

    for (int kt = 0; kt < ktiles; ++kt) {
#if USE_ASYNC_LDS
        wait_async0();                   // this wave's async loads for `buf` done
#endif
        __syncthreads();                 // whole tile visible; prev compute done
        const int buf = kt & 1;

        if (kt + 1 < ktiles) {           // kick next K tile into other buffer
            const int k0 = (kt + 1) * TK;
            const int nb = buf ^ 1;
#if USE_ASYNC_LDS
#pragma unroll
            for (int c = tid; c < (TM * TK / 8); c += 256) {
                const int row = c >> 2, col = (c & 3) * 8;
                async_copy16(&A[(size_t)(mblk + row) * Kpad + k0 + col],
                             &As[nb][row][col]);
            }
            {
                const int row = tid >> 2, col = (tid & 3) * 8;
                async_copy16(&W[(size_t)(nblk + row) * Kpad + k0 + col],
                             &Ws[nb][row][col]);
            }
#else
            __builtin_prefetch(&A[(size_t)mblk * Kpad + k0], 0, 0);
#pragma unroll
            for (int c = tid; c < (TM * TK / 8); c += 256) {
                const int row = c >> 2, col = (c & 3) * 8;
                *(v8h*)&As[nb][row][col] =
                    *(const v8h*)&A[(size_t)(mblk + row) * Kpad + k0 + col];
            }
            {
                const int row = tid >> 2, col = (tid & 3) * 8;
                *(v8h*)&Ws[nb][row][col] =
                    *(const v8h*)&W[(size_t)(nblk + row) * Kpad + k0 + col];
            }
#endif
        }

        // Fragments per ISA 7.12.2 layouts (wave32, 16-bit A 16x32 / B 32x16).
        V16 a0, a1, b0, b1;
        a0.h[0] = *(const v8h*)&As[buf][wm * 32 + r     ][     kg * 8];
        a0.h[1] = *(const v8h*)&As[buf][wm * 32 + r     ][16 + kg * 8];
        a1.h[0] = *(const v8h*)&As[buf][wm * 32 + 16 + r][     kg * 8];
        a1.h[1] = *(const v8h*)&As[buf][wm * 32 + 16 + r][16 + kg * 8];
        b0.h[0] = *(const v8h*)&Ws[buf][wn * 32 + r     ][     kg * 8];
        b0.h[1] = *(const v8h*)&Ws[buf][wn * 32 + r     ][16 + kg * 8];
        b1.h[0] = *(const v8h*)&Ws[buf][wn * 32 + 16 + r][     kg * 8];
        b1.h[1] = *(const v8h*)&Ws[buf][wn * 32 + 16 + r][16 + kg * 8];

        acc[0][0] = __builtin_amdgcn_wmma_f32_16x16x32_f16(false, a0.v, false, b0.v,
                                                           (short)0, acc[0][0], false, false);
        acc[0][1] = __builtin_amdgcn_wmma_f32_16x16x32_f16(false, a0.v, false, b1.v,
                                                           (short)0, acc[0][1], false, false);
        acc[1][0] = __builtin_amdgcn_wmma_f32_16x16x32_f16(false, a1.v, false, b0.v,
                                                           (short)0, acc[1][0], false, false);
        acc[1][1] = __builtin_amdgcn_wmma_f32_16x16x32_f16(false, a1.v, false, b1.v,
                                                           (short)0, acc[1][1], false, false);
    }

    // C/D layout: element t of v8f is row (t + 8*kg) of the 16x16 tile, col = r.
#pragma unroll
    for (int i = 0; i < 2; ++i)
#pragma unroll
        for (int j = 0; j < 2; ++j) {
            const int col  = nblk + wn * 32 + j * 16 + r;
            const int row0 = mblk + wm * 32 + i * 16 + kg * 8;
            const float bv = (col < biasN) ? bias[col] : 0.0f;
#pragma unroll
            for (int t = 0; t < 8; ++t)
                C[(size_t)(row0 + t) * Npad + col] = acc[i][j][t] + bv;
        }
}

// --------------------------- column statistics -----------------------------
__global__ void colstats_partial(const float* __restrict__ Z, int Npad,
                                 float* __restrict__ psum, float* __restrict__ psq)
{
    const int col = blockIdx.x * blockDim.x + threadIdx.x;
    if (col >= Npad) return;
    const int rows = MROWS / 32;
    const int r0 = blockIdx.y * rows;
    float s = 0.f, q = 0.f;
    for (int rr = 0; rr < rows; ++rr) {
        const float v = Z[(size_t)(r0 + rr) * Npad + col];
        s += v; q += v * v;
    }
    psum[(size_t)blockIdx.y * Npad + col] = s;
    psq [(size_t)blockIdx.y * Npad + col] = q;
}

__global__ void colstats_final(const float* __restrict__ psum, const float* __restrict__ psq,
                               int Npad, float* __restrict__ mean, float* __restrict__ var)
{
    const int col = blockIdx.x * blockDim.x + threadIdx.x;
    if (col >= Npad) return;
    float s = 0.f, q = 0.f;
    for (int p = 0; p < 32; ++p) { s += psum[(size_t)p * Npad + col]; q += psq[(size_t)p * Npad + col]; }
    const float mu = s / (float)MROWS;
    mean[col] = mu;
    var[col]  = q / (float)MROWS - mu * mu;   // biased variance (matches jnp.var)
}

// ------------------------------- RNG (N(0,1)) ------------------------------
__device__ __forceinline__ uint32_t hash32(uint32_t x)
{
    x ^= x >> 16; x *= 0x7feb352dU;
    x ^= x >> 15; x *= 0x846ca68bU;
    x ^= x >> 16; return x;
}
__device__ __forceinline__ float gauss01(uint32_t seed, uint32_t idx)
{
    const uint32_t u1 = hash32(idx * 2u + 1u + seed * 0x9E3779B9u);
    const uint32_t u2 = hash32(idx * 2u + 2u + seed * 0x85EBCA6BU);
    const float f1 = ((u1 >> 8) + 1.0f) * 0x1.0p-24f;   // (0,1]
    const float f2 = (u2 >> 8) * 0x1.0p-24f;            // [0,1)
    return sqrtf(-2.0f * __logf(f1)) * __cosf(6.28318530718f * f2);
}

// ---------------- fused BN + noise + activation (encoder) ------------------
__global__ void bn_act(const float* __restrict__ Z,
                       const float* __restrict__ mean, const float* __restrict__ var,
                       const float* __restrict__ beta, const float* __restrict__ gamma,
                       float noise_std, uint32_t seed,
                       float* __restrict__ zout,        // nullable, padded stride
                       _Float16* __restrict__ Hf16,     // padded stride
                       float* __restrict__ Hf32,        // nullable, padded stride
                       int Npad, int N)
{
    const size_t idx = (size_t)blockIdx.x * blockDim.x + threadIdx.x;
    const int col = (int)(idx % Npad);
    const int row = (int)(idx / Npad);
    float h = 0.f, z = 0.f;
    if (col < N) {
        z = (Z[idx] - mean[col]) * rsqrtf(var[col] + 1e-5f);
        if (noise_std > 0.f) z += noise_std * gauss01(seed, (uint32_t)(row * N + col));
        const float zb = z + beta[col];
        h = gamma ? gamma[col] * zb : fmaxf(zb, 0.f);
    }
    if (zout) zout[idx] = (col < N) ? z : 0.f;
    Hf16[idx] = (_Float16)h;
    if (Hf32) Hf32[idx] = h;
}

// ---------------------- input prep (clean + noisy x) -----------------------
__global__ void input_prep(const float* __restrict__ x, int K, int Kpad, uint32_t seed,
                           float* __restrict__ zn0, _Float16* __restrict__ Hn16,
                           _Float16* __restrict__ Hc16)
{
    const size_t idx = (size_t)blockIdx.x * blockDim.x + threadIdx.x;
    const int col = (int)(idx % Kpad);
    const int row = (int)(idx / Kpad);
    float v = 0.f, zn = 0.f;
    if (col < K) {
        v  = x[(size_t)row * K + col];
        zn = v + gauss01(seed, (uint32_t)(row * K + col));   // input noise std = 1
    }
    zn0[idx]  = zn;
    Hn16[idx] = (_Float16)zn;
    Hc16[idx] = (_Float16)v;
}

// ------------------------------- log_softmax -------------------------------
__global__ void logsoftmax10(const float* __restrict__ H, float* __restrict__ y)
{
    const int row = blockIdx.x * blockDim.x + threadIdx.x;
    if (row >= MROWS) return;
    const float* h = &H[(size_t)row * 64];
    float mx = h[0];
#pragma unroll
    for (int j = 1; j < 10; ++j) mx = fmaxf(mx, h[j]);
    float s = 0.f;
#pragma unroll
    for (int j = 0; j < 10; ++j) s += __expf(h[j] - mx);
    const float lse = mx + __logf(s);
#pragma unroll
    for (int j = 0; j < 10; ++j) y[(size_t)row * 10 + j] = h[j] - lse;
}

// ----------------------- decoder combinator g(u, zn) -----------------------
__device__ __forceinline__ float sigf(float x) { return 1.0f / (1.0f + __expf(-x)); }

__global__ void combine(const float* __restrict__ U,
                        const float* __restrict__ mean, const float* __restrict__ var,
                        const float* __restrict__ alpha,   // [10][N]
                        const float* __restrict__ zn,      // padded stride Npad
                        int Npad, int N,
                        _Float16* __restrict__ zh16,       // nullable, padded stride
                        float* __restrict__ zh32)          // nullable, stride N (d_out)
{
    const size_t idx = (size_t)blockIdx.x * blockDim.x + threadIdx.x;
    const int col = (int)(idx % Npad);
    const int row = (int)(idx / Npad);
    float zh = 0.f;
    if (col < N) {
        const float u  = (U[idx] - mean[col]) * rsqrtf(var[col] + 1e-5f);
        const float a0 = alpha[0 * N + col], a1 = alpha[1 * N + col], a2 = alpha[2 * N + col];
        const float a3 = alpha[3 * N + col], a4 = alpha[4 * N + col], a5 = alpha[5 * N + col];
        const float a6 = alpha[6 * N + col], a7 = alpha[7 * N + col], a8 = alpha[8 * N + col];
        const float a9 = alpha[9 * N + col];
        const float mu = a0 * sigf(a1 * u + a2) + a3 * u + a4;
        const float vv = a5 * sigf(a6 * u + a7) + a8 * u + a9;
        zh = (zn[idx] - mu) * vv + mu;
        if (zh32) zh32[(size_t)row * N + col] = zh;
    }
    if (zh16) zh16[idx] = (_Float16)zh;
}

// -------------------- weight f32 -> padded f16 conversion ------------------
__global__ void cvt_w(const float* __restrict__ W, int N, int K,
                      _Float16* __restrict__ Wp, int Npad, int Kpad)
{
    const size_t idx = (size_t)blockIdx.x * blockDim.x + threadIdx.x;
    if (idx >= (size_t)Npad * Kpad) return;
    const int rr = (int)(idx / Kpad), cc = (int)(idx % Kpad);
    Wp[idx] = (rr < N && cc < K) ? (_Float16)W[(size_t)rr * K + cc] : (_Float16)0.f;
}

// ================================ host side ================================

extern "C" void kernel_launch(void* const* d_in, const int* in_sizes, int n_in,
                              void* d_out, int out_size, void* d_ws, size_t ws_size,
                              hipStream_t stream)
{
    (void)in_sizes; (void)n_in; (void)out_size; (void)ws_size;

    static const int SZ[7] = {784, 1000, 500, 250, 250, 250, 10};
    static const int PD[7] = {832, 1024, 512, 256, 256, 256, 64};   // pad to mult of 64
    const int M = MROWS;

    // input dict order: x, enc_W[6], enc_b[6], dec_W[6], dec_b[6], alphas[7], gamma, beta[7]
    const float* x = (const float*)d_in[0];
    const float* encW[6]; const float* encB[6];
    const float* decW[6]; const float* decB[6];
    const float* alph[7]; const float* beta[7];
    for (int i = 0; i < 6; ++i) { encW[i] = (const float*)d_in[1 + i];  encB[i] = (const float*)d_in[7 + i]; }
    for (int i = 0; i < 6; ++i) { decW[i] = (const float*)d_in[13 + i]; decB[i] = (const float*)d_in[19 + i]; }
    for (int l = 0; l < 7; ++l) { alph[l] = (const float*)d_in[25 + l]; beta[l] = (const float*)d_in[33 + l]; }
    const float* gamma = (const float*)d_in[32];

    // --- workspace bump allocator ---
    char* base = (char*)d_ws; size_t off = 0;
    auto alloc = [&](size_t bytes) -> void* {
        off = (off + 255) & ~(size_t)255;
        void* p = base + off; off += bytes; return p;
    };

    _Float16* Wenc[6]; _Float16* Wdec[6];
    for (int i = 0; i < 6; ++i) Wenc[i] = (_Float16*)alloc((size_t)PD[i + 1] * PD[i] * 2);
    for (int l = 0; l < 6; ++l) Wdec[l] = (_Float16*)alloc((size_t)PD[l] * PD[l + 1] * 2);
    float* zn[7];
    for (int l = 0; l < 7; ++l) zn[l] = (float*)alloc((size_t)M * PD[l] * 4);
    float* Zscr   = (float*)alloc((size_t)M * 1024 * 4);
    float* meanB  = (float*)alloc(1024 * 4);
    float* varB   = (float*)alloc(1024 * 4);
    float* psum   = (float*)alloc((size_t)32 * 1024 * 4);
    float* psq    = (float*)alloc((size_t)32 * 1024 * 4);
    _Float16* B0  = (_Float16*)alloc((size_t)M * 1024 * 2);   // clean ping / zhat ping
    _Float16* B1  = (_Float16*)alloc((size_t)M * 1024 * 2);   // clean pong / zhat pong
    _Float16* B2  = (_Float16*)alloc((size_t)M * 1024 * 2);   // noisy ping
    _Float16* B3  = (_Float16*)alloc((size_t)M * 1024 * 2);   // noisy pong
    float* HcLast = (float*)alloc((size_t)M * 64 * 4);
    float* HnLast = (float*)alloc((size_t)M * 64 * 4);

    float* y_out    = (float*)d_out;
    float* zhat_out = (float*)d_out + (size_t)M * 10;

    auto run_stats = [&](const float* Z, int Npad) {
        dim3 g1((Npad + 255) / 256, 32);
        colstats_partial<<<g1, 256, 0, stream>>>(Z, Npad, psum, psq);
        colstats_final<<<dim3((Npad + 255) / 256), 256, 0, stream>>>(psum, psq, Npad, meanB, varB);
    };
    auto run_gemm = [&](const _Float16* A, const _Float16* W, const float* bias, int biasN,
                        float* C, int Kpad, int Npad) {
        gemm_wmma_f16<<<dim3(M / TM, Npad / TN), 256, 0, stream>>>(A, W, bias, biasN, C, Kpad, Npad);
    };

    // --- weight conversion ---
    for (int i = 0; i < 6; ++i) {
        const size_t n = (size_t)PD[i + 1] * PD[i];
        cvt_w<<<dim3((unsigned)((n + 255) / 256)), 256, 0, stream>>>(encW[i], SZ[i + 1], SZ[i],
                                                                     Wenc[i], PD[i + 1], PD[i]);
    }
    for (int l = 0; l < 6; ++l) {
        const size_t n = (size_t)PD[l] * PD[l + 1];
        cvt_w<<<dim3((unsigned)((n + 255) / 256)), 256, 0, stream>>>(decW[l], SZ[l], SZ[l + 1],
                                                                     Wdec[l], PD[l], PD[l + 1]);
    }

    // --- input prep: z_noise[0] = x + N(0,1); f16 copies for both paths ---
    input_prep<<<dim3((unsigned)((size_t)M * PD[0] / 256)), 256, 0, stream>>>(
        x, SZ[0], PD[0], 0xC0FFEEu, zn[0], B2, B0);

    // --- clean encoder ---
    {
        _Float16 *cur = B0, *nxt = B1;
        for (int i = 1; i <= 6; ++i) {
            run_gemm(cur, Wenc[i - 1], encB[i - 1], SZ[i], Zscr, PD[i - 1], PD[i]);
            run_stats(Zscr, PD[i]);
            bn_act<<<dim3((unsigned)((size_t)M * PD[i] / 256)), 256, 0, stream>>>(
                Zscr, meanB, varB, beta[i], (i == 6) ? gamma : (const float*)nullptr,
                0.0f, 0u, (float*)nullptr, nxt, (i == 6) ? HcLast : (float*)nullptr,
                PD[i], SZ[i]);
            _Float16* t = cur; cur = nxt; nxt = t;
        }
        logsoftmax10<<<dim3(M / 256), 256, 0, stream>>>(HcLast, y_out);
    }

    // --- noisy encoder (stores z_noise[i]) ---
    {
        _Float16 *cur = B2, *nxt = B3;
        for (int i = 1; i <= 6; ++i) {
            run_gemm(cur, Wenc[i - 1], encB[i - 1], SZ[i], Zscr, PD[i - 1], PD[i]);
            run_stats(Zscr, PD[i]);
            bn_act<<<dim3((unsigned)((size_t)M * PD[i] / 256)), 256, 0, stream>>>(
                Zscr, meanB, varB, beta[i], (i == 6) ? gamma : (const float*)nullptr,
                0.2f, 0xABC0u + (uint32_t)i, zn[i], nxt, (i == 6) ? HnLast : (float*)nullptr,
                PD[i], SZ[i]);
            _Float16* t = cur; cur = nxt; nxt = t;
        }
    }

    // --- decoder (top: u = BN(h_n)) ---
    run_stats(HnLast, PD[6]);
    combine<<<dim3((unsigned)((size_t)M * PD[6] / 256)), 256, 0, stream>>>(
        HnLast, meanB, varB, alph[6], zn[6], PD[6], SZ[6], B0, (float*)nullptr);

    {
        _Float16 *zcur = B0, *znxt = B1;
        for (int l = 5; l >= 0; --l) {
            run_gemm(zcur, Wdec[l], decB[l], SZ[l], Zscr, PD[l + 1], PD[l]);
            run_stats(Zscr, PD[l]);
            combine<<<dim3((unsigned)((size_t)M * PD[l] / 256)), 256, 0, stream>>>(
                Zscr, meanB, varB, alph[l], zn[l], PD[l], SZ[l],
                (l > 0) ? znxt : (_Float16*)nullptr,
                (l == 0) ? zhat_out : (float*)nullptr);
            _Float16* t = zcur; zcur = znxt; znxt = t;
        }
    }
}
